// DoshaGAT_9749575762552
// MI455X (gfx1250) — matrix operand order repaired and checked by reading, not compiled, not run on hardware
//
#include <hip/hip_runtime.h>
#include <hip/hip_bf16.h>
#include <cstdint>

// ---------------------------------------------------------------------------
// 2-layer GAT for MI455X (gfx1250, wave32).
//  GEMMs via v_wmma_f32_16x16x32_bf16 (fp32 accumulate), edge softmax via
//  ordered-uint float atomicMax + fp32 atomicAdd (L2-resident working set).
//  Workspace usage: ~132.1 MB (see offsets below).
// ---------------------------------------------------------------------------

#define GN    50000
#define GE    800000
#define GEP   850000          // E + N self loops
#define INC   128
#define HID   64
#define HEADS 4
#define C1    256             // HEADS*HID
#define OUTC  16
#define SLOPE 0.2f

#define CDIV(a, b) (((a) + (b) - 1) / (b))

typedef __attribute__((ext_vector_type(16))) __bf16 v16bf;
typedef __attribute__((ext_vector_type(8)))  float  v8f;

union BF16x16 { unsigned short u[16]; v16bf v; };

// ---- helpers --------------------------------------------------------------

__device__ __forceinline__ unsigned short f2bf(float f) {
  unsigned int u = __float_as_uint(f);
  u += 0x7FFFu + ((u >> 16) & 1u);          // round-to-nearest-even
  return (unsigned short)(u >> 16);
}

// order-preserving float<->uint encoding for atomicMax over signed floats
__device__ __forceinline__ unsigned int encf(float f) {
  unsigned int u = __float_as_uint(f);
  return (u & 0x80000000u) ? ~u : (u | 0x80000000u);
}
__device__ __forceinline__ float decf(unsigned int k) {
  return __uint_as_float((k & 0x80000000u) ? (k & 0x7FFFFFFFu) : ~k);
}

__device__ __forceinline__ void edge_nodes(const long long* __restrict__ ei,
                                           long long e, long long& s, long long& d) {
  if (e < (long long)GE) { s = ei[e]; d = ei[(long long)GE + e]; }
  else                   { s = d = e - (long long)GE; }
}

// ---- precision conversion / layout kernels --------------------------------

__global__ void k_cvt_bf16(const float* __restrict__ x, unsigned short* __restrict__ xb, int n) {
  int i = blockIdx.x * blockDim.x + threadIdx.x;
  if (i < n) xb[i] = f2bf(x[i]);
}

// wt[n*K + k] = bf16(w[k*Nn + n])   (transpose so B fragments are contiguous)
__global__ void k_cvt_w_t(const float* __restrict__ w, unsigned short* __restrict__ wt,
                          int K, int Nn) {
  int i = blockIdx.x * blockDim.x + threadIdx.x;
  if (i >= K * Nn) return;
  int n = i / K, k = i % K;
  wt[i] = f2bf(w[(size_t)k * Nn + n]);
}

// ---- WMMA GEMM 1: h1[N,256] = x[N,128] @ W1[128,256] ----------------------

__global__ __launch_bounds__(128)
void k_gemm1_wmma(const unsigned short* __restrict__ xb,
                  const unsigned short* __restrict__ w1t,
                  float* __restrict__ h1) {
  const int wave = threadIdx.x >> 5;
  const int lane = threadIdx.x & 31;
  const int lrow = lane & 15;
  const int lhi  = lane >> 4;
  const int m0   = blockIdx.x * 16;

  // preload the 4 A fragments covering K=128 (contiguous k = 16*lhi + e, matching B)
  BF16x16 a[4];
  const unsigned short* xrow = xb + (size_t)(m0 + lrow) * INC;
#pragma unroll
  for (int kk = 0; kk < 4; ++kk) {
    const int kb = kk * 32;
#pragma unroll
    for (int e = 0; e < 16; ++e) {
      a[kk].u[e] = xrow[kb + lhi * 16 + e];
    }
  }

#pragma unroll
  for (int nt = 0; nt < 4; ++nt) {
    const int n0 = (wave + nt * 4) * 16;
    v8f acc = {};
#pragma unroll
    for (int kk = 0; kk < 4; ++kk) {
      BF16x16 b;
      const unsigned short* wr = w1t + (size_t)(n0 + lrow) * INC + kk * 32 + lhi * 16;
#pragma unroll
      for (int e = 0; e < 16; ++e) b.u[e] = wr[e];
      acc = __builtin_amdgcn_wmma_f32_16x16x32_bf16(false, a[kk].v, false, b.v,
                                                    (short)0, acc, false, false);
    }
    float* hrow = h1 + (size_t)(m0 + lhi * 8) * C1 + n0 + lrow;
#pragma unroll
    for (int v = 0; v < 8; ++v) hrow[(size_t)v * C1] = acc[v];
  }
}

// ---- WMMA GEMM 2: h2[N,16] = elu(h1+b1)_bf16[N,256] @ W2[256,16] ----------

__global__ __launch_bounds__(128)
void k_gemm2_wmma(const unsigned short* __restrict__ hb,
                  const unsigned short* __restrict__ w2t,
                  float* __restrict__ h2) {
  const int wave  = threadIdx.x >> 5;
  const int lane  = threadIdx.x & 31;
  const int lrow  = lane & 15;
  const int lhi   = lane >> 4;
  const int mtile = blockIdx.x * 4 + wave;
  if (mtile >= GN / 16) return;
  const int m0 = mtile * 16;

  v8f acc = {};
  const unsigned short* hrow = hb + (size_t)(m0 + lrow) * C1;
  const unsigned short* wr0  = w2t + (size_t)lrow * C1;
#pragma unroll
  for (int kk = 0; kk < 8; ++kk) {
    const int kb = kk * 32;
    BF16x16 a, b;
#pragma unroll
    for (int e = 0; e < 16; ++e) {
      a.u[e] = hrow[kb + lhi * 16 + e];
    }
#pragma unroll
    for (int e = 0; e < 16; ++e) b.u[e] = wr0[kb + lhi * 16 + e];
    acc = __builtin_amdgcn_wmma_f32_16x16x32_bf16(false, a.v, false, b.v,
                                                  (short)0, acc, false, false);
  }
  float* out = h2 + (size_t)(m0 + lhi * 8) * OUTC + lrow;
#pragma unroll
  for (int v = 0; v < 8; ++v) out[(size_t)v * OUTC] = acc[v];
}

// ---- attention coefficients ----------------------------------------------

__global__ void k_alpha(const float* __restrict__ h, const float* __restrict__ a_src,
                        const float* __restrict__ a_dst, float* __restrict__ asv,
                        float* __restrict__ adv, int H, int F) {
  int i = blockIdx.x * blockDim.x + threadIdx.x;
  if (i >= GN * H) return;
  int n = i / H, hh = i % H;
  const float* hr = h + (size_t)n * H * F + (size_t)hh * F;
  const float* as = a_src + hh * F;
  const float* ad = a_dst + hh * F;
  float s = 0.f, d = 0.f;
  for (int f = 0; f < F; ++f) { float v = hr[f]; s = fmaf(v, as[f], s); d = fmaf(v, ad[f], d); }
  asv[i] = s; adv[i] = d;
}

// ---- segment-softmax edge passes ------------------------------------------

__global__ void k_edge_max(const long long* __restrict__ ei, const float* __restrict__ asv,
                           const float* __restrict__ adv, unsigned int* __restrict__ mkey,
                           int H, int total) {
  int i = blockIdx.x * blockDim.x + threadIdx.x;
  if (i >= total) return;
  int h = i % H; long long e = i / H, s, d;
  edge_nodes(ei, e, s, d);
  float sc = asv[s * H + h] + adv[d * H + h];
  sc = sc > 0.f ? sc : SLOPE * sc;
  atomicMax(&mkey[d * H + h], encf(sc));
}

__global__ void k_edge_exp(const long long* __restrict__ ei, const float* __restrict__ asv,
                           const float* __restrict__ adv, const unsigned int* __restrict__ mkey,
                           float* __restrict__ ex, float* __restrict__ denom,
                           int H, int total) {
  int i = blockIdx.x * blockDim.x + threadIdx.x;
  if (i >= total) return;
  int h = i % H; long long e = i / H, s, d;
  edge_nodes(ei, e, s, d);
  float sc = asv[s * H + h] + adv[d * H + h];
  sc = sc > 0.f ? sc : SLOPE * sc;
  float v = expf(sc - decf(mkey[d * H + h]));
  ex[i] = v;
  atomicAdd(&denom[d * H + h], v);
}

__global__ void k_edge_norm(const long long* __restrict__ ei, float* __restrict__ ex,
                            const float* __restrict__ denom, int H, int total) {
  int i = blockIdx.x * blockDim.x + threadIdx.x;
  if (i >= total) return;
  int h = i % H; long long e = i / H, s, d;
  edge_nodes(ei, e, s, d);
  ex[i] = ex[i] / denom[d * H + h];
}

__global__ void k_edge_agg(const long long* __restrict__ ei, const float* __restrict__ alpha,
                           const float* __restrict__ hf, float* __restrict__ out,
                           int H, int F, int total) {
  int i = blockIdx.x * blockDim.x + threadIdx.x;
  if (i >= total) return;                      // total = EP * F
  int f = i % F; long long e = i / F, s, d;
  edge_nodes(ei, e, s, d);
  const int C = H * F;
  for (int h = 0; h < H; ++h) {
    float al = alpha[e * H + h];
    atomicAdd(&out[d * C + h * F + f], al * hf[s * C + h * F + f]);
  }
}

// ---- bias + ELU + bf16 re-quantize for layer-2 GEMM -----------------------

__global__ void k_bias_elu_bf16(const float* __restrict__ agg, const float* __restrict__ b,
                                unsigned short* __restrict__ hb) {
  int i = blockIdx.x * blockDim.x + threadIdx.x;
  if (i >= GN * C1) return;
  float v = agg[i] + b[i % C1];
  v = v > 0.f ? v : (expf(v) - 1.f);
  hb[i] = f2bf(v);
}

// ---- final bias + log_softmax --------------------------------------------

__global__ void k_logsoftmax(const float* __restrict__ agg2, const float* __restrict__ b2,
                             float* __restrict__ out) {
  int n = blockIdx.x * blockDim.x + threadIdx.x;
  if (n >= GN) return;
  float v[OUTC], mx = -3.4e38f;
  for (int c = 0; c < OUTC; ++c) { v[c] = agg2[(size_t)n * OUTC + c] + b2[c]; mx = fmaxf(mx, v[c]); }
  float s = 0.f;
  for (int c = 0; c < OUTC; ++c) s += expf(v[c] - mx);
  float l = logf(s);
  for (int c = 0; c < OUTC; ++c) out[(size_t)n * OUTC + c] = v[c] - mx - l;
}

// ---------------------------------------------------------------------------

extern "C" void kernel_launch(void* const* d_in, const int* in_sizes, int n_in,
                              void* d_out, int out_size, void* d_ws, size_t ws_size,
                              hipStream_t stream) {
  const float*     x     = (const float*)d_in[0];
  const long long* ei    = (const long long*)d_in[1];   // int64 [2,E]
  const float*     W1    = (const float*)d_in[2];
  const float*     asrc1 = (const float*)d_in[3];
  const float*     adst1 = (const float*)d_in[4];
  const float*     b1    = (const float*)d_in[5];
  const float*     W2    = (const float*)d_in[6];
  const float*     asrc2 = (const float*)d_in[7];
  const float*     adst2 = (const float*)d_in[8];
  const float*     b2    = (const float*)d_in[9];
  float*           out   = (float*)d_out;

  char* ws = (char*)d_ws;
  // workspace layout (bytes) -- total 132,073,728
  float*          h1    = (float*)(ws + 0);                 // N*256 f32 (51.2e6)
  unsigned short* h1b   = (unsigned short*)(ws + 0);        // reuse after agg (25.6e6)
  float*          out1  = (float*)(ws + 51200000);          // N*256 f32
  unsigned short* xb    = (unsigned short*)(ws + 102400000);// N*128 bf16 (dead after gemm1)
  float*          h2    = (float*)(ws + 102400000);         // N*16 f32 (reuse xb region)
  float*          agg2  = (float*)(ws + 105600000);         // N*16 f32
  float*          asv2  = (float*)(ws + 108800000);         // N f32
  float*          adv2  = (float*)(ws + 109000000);
  unsigned int*   mk2   = (unsigned int*)(ws + 109200000);
  float*          den2  = (float*)(ws + 109400000);
  float*          exp1  = (float*)(ws + 115200000);         // EP*4 f32 (13.6e6); exp2 reuses
  float*          exp2  = (float*)(ws + 115200000);
  float*          asv1  = (float*)(ws + 128800000);         // N*4 f32
  float*          adv1  = (float*)(ws + 129600000);
  unsigned int*   mk1   = (unsigned int*)(ws + 130400000);
  float*          den1  = (float*)(ws + 131200000);
  unsigned short* w1t   = (unsigned short*)(ws + 132000000);// 256x128 bf16
  unsigned short* w2t   = (unsigned short*)(ws + 132065536);// 16x256 bf16

  const int B = 256;

  // zero accumulators (key 0 sorts below encf of every finite score -> valid -inf init;
  // every node has a self-loop so every slot is touched)
  hipMemsetAsync(mk1,  0, (size_t)GN * HEADS * 4, stream);
  hipMemsetAsync(den1, 0, (size_t)GN * HEADS * 4, stream);
  hipMemsetAsync(out1, 0, (size_t)GN * C1 * 4, stream);
  hipMemsetAsync(mk2,  0, (size_t)GN * 4, stream);
  hipMemsetAsync(den2, 0, (size_t)GN * 4, stream);
  hipMemsetAsync(agg2, 0, (size_t)GN * OUTC * 4, stream);

  // ---- layer 1 ----
  k_cvt_bf16<<<CDIV(GN * INC, B), B, 0, stream>>>(x, xb, GN * INC);
  k_cvt_w_t <<<CDIV(INC * C1, B), B, 0, stream>>>(W1, w1t, INC, C1);
  k_cvt_w_t <<<CDIV(C1 * OUTC, B), B, 0, stream>>>(W2, w2t, C1, OUTC);

  k_gemm1_wmma<<<GN / 16, 128, 0, stream>>>(xb, w1t, h1);

  k_alpha<<<CDIV(GN * HEADS, B), B, 0, stream>>>(h1, asrc1, adst1, asv1, adv1, HEADS, HID);

  const int t1 = GEP * HEADS;
  k_edge_max <<<CDIV(t1, B), B, 0, stream>>>(ei, asv1, adv1, mk1, HEADS, t1);
  k_edge_exp <<<CDIV(t1, B), B, 0, stream>>>(ei, asv1, adv1, mk1, exp1, den1, HEADS, t1);
  k_edge_norm<<<CDIV(t1, B), B, 0, stream>>>(ei, exp1, den1, HEADS, t1);
  k_edge_agg <<<CDIV(GEP * HID, B), B, 0, stream>>>(ei, exp1, h1, out1, HEADS, HID, GEP * HID);

  k_bias_elu_bf16<<<CDIV(GN * C1, B), B, 0, stream>>>(out1, b1, h1b);

  // ---- layer 2 ----
  k_gemm2_wmma<<<CDIV(GN / 16, 4), 128, 0, stream>>>(h1b, w2t, h2);

  k_alpha<<<CDIV(GN, B), B, 0, stream>>>(h2, asrc2, adst2, asv2, adv2, 1, OUTC);

  const int t2 = GEP;
  k_edge_max <<<CDIV(t2, B), B, 0, stream>>>(ei, asv2, adv2, mk2, 1, t2);
  k_edge_exp <<<CDIV(t2, B), B, 0, stream>>>(ei, asv2, adv2, mk2, exp2, den2, 1, t2);
  k_edge_norm<<<CDIV(t2, B), B, 0, stream>>>(ei, exp2, den2, 1, t2);
  k_edge_agg <<<CDIV(GEP * OUTC, B), B, 0, stream>>>(ei, exp2, h2, agg2, 1, OUTC, GEP * OUTC);

  k_logsoftmax<<<CDIV(GN, B), B, 0, stream>>>(agg2, b2, out);
}
